// LSTMModel_40278203302621
// MI455X (gfx1250) — matrix operand order repaired and checked by reading, not compiled, not run on hardware
//
#include <hip/hip_runtime.h>
#include <hip/hip_bf16.h>

// ---------------------------------------------------------------------------
// Persistent 2-layer LSTM + autoregressive rollout for MI455X (gfx1250).
// 32 workgroups x 16 batch rows, zero inter-WG sync, weights fragment-packed
// in LDS, recurrence driven by v_wmma_f32_16x16x32_bf16.
// Round 3: scalar-uniform tile scheduling (no EXEC-predicated WMMA), bias
// seeded into the WMMA C operand, immediate-offset gate stores.
// ---------------------------------------------------------------------------

typedef __attribute__((ext_vector_type(16))) __bf16 v16bf;
typedef __attribute__((ext_vector_type(8)))  float  v8f;

struct alignas(16) H8 { __bf16 h[8]; };

#define HDIM      100
#define SEQL      168
#define BATCH     512
#define NSTEPS    48
#define ROWS      16          // batch rows per workgroup
#define NGATE     400         // 4*H
#define NTILES    25          // 400/16
#define K1PAD     224         // layer-1 K padded (200 -> 224); layer-0 uses first 128
#define NC0       4           // K chunks layer 0
#define NC1       7           // K chunks layer 1

// ---- LDS layout (bytes) ----
#define OFF_W0    0                         // 25*4*512 bf16 = 102400
#define OFF_W1    102400                    // 25*7*512 bf16 = 179200
#define OFF_HA    281600                    // 16*224 bf16   = 7168
#define OFF_XS    288768                    // 16*168 bf16   = 5376
#define OFF_G     294144                    // 16*400 f32    = 25600
#define OFF_B0    319744                    // 400 f32       = 1600
#define OFF_B1    321344                    // 400 f32       = 1600
#define OFF_WI    322944                    // 400 f32       = 1600
#define OFF_FC    324544                    // 100 f32       = 400
#define OFF_YV    324944                    // 16 f32        = 64
#define SMEM_BYTES 325008

__device__ __forceinline__ float fast_sig(float x) {
  return __builtin_amdgcn_rcpf(1.0f + __builtin_amdgcn_exp2f(-1.44269504f * x));
}
__device__ __forceinline__ float fast_tanh(float x) {
  return 2.0f * __builtin_amdgcn_rcpf(1.0f + __builtin_amdgcn_exp2f(-2.88539008f * x)) - 1.0f;
}

__device__ __forceinline__ v16bf load_frag_pair(const H8* p0, const H8* p1) {
  H8 lo = *p0, hi = *p1;
  v16bf v;
#pragma unroll
  for (int e = 0; e < 8; ++e) { v[e] = lo.h[e]; v[e + 8] = hi.h[e]; }
  return v;
}

// Two N-tiles with independent accumulator chains (hides WMMA latency).
template <int NC>
__device__ __forceinline__ void gemm_pair(const __bf16* __restrict__ Wf,
                                          const v16bf* __restrict__ a,
                                          const float* __restrict__ bias,
                                          float* __restrict__ gates,
                                          int lane, int nt0, int nt1) {
  const int am = lane & 15;
  const int m0 = (lane & 16) ? 8 : 0;
  const int col0 = nt0 * 16 + am, col1 = nt1 * 16 + am;
  const float bv0 = bias[col0], bv1 = bias[col1];
  v8f acc0, acc1;
#pragma unroll
  for (int r = 0; r < 8; ++r) { acc0[r] = bv0; acc1[r] = bv1; }
#pragma unroll
  for (int kc = 0; kc < NC; ++kc) {
    const H8* pb0 = reinterpret_cast<const H8*>(Wf + ((nt0 * NC + kc) << 9) + (lane << 4));
    const H8* pb1 = reinterpret_cast<const H8*>(Wf + ((nt1 * NC + kc) << 9) + (lane << 4));
    v16bf b0 = load_frag_pair(pb0, pb0 + 1);
    v16bf b1 = load_frag_pair(pb1, pb1 + 1);
    acc0 = __builtin_amdgcn_wmma_f32_16x16x32_bf16(false, a[kc], false, b0,
                                                   (short)0, acc0, false, false);
    acc1 = __builtin_amdgcn_wmma_f32_16x16x32_bf16(false, a[kc], false, b1,
                                                   (short)0, acc1, false, false);
  }
  float* g0 = gates + m0 * NGATE + col0;
  float* g1 = gates + m0 * NGATE + col1;
#pragma unroll
  for (int r = 0; r < 8; ++r) { g0[r * NGATE] = acc0[r]; g1[r * NGATE] = acc1[r]; }
}

template <int NC>
__device__ __forceinline__ void gemm_single(const __bf16* __restrict__ Wf,
                                            const v16bf* __restrict__ a,
                                            const float* __restrict__ bias,
                                            float* __restrict__ gates,
                                            int lane, int nt0) {
  const int am = lane & 15;
  const int m0 = (lane & 16) ? 8 : 0;
  const int col0 = nt0 * 16 + am;
  const float bv0 = bias[col0];
  v8f acc0;
#pragma unroll
  for (int r = 0; r < 8; ++r) acc0[r] = bv0;
#pragma unroll
  for (int kc = 0; kc < NC; ++kc) {
    const H8* pb0 = reinterpret_cast<const H8*>(Wf + ((nt0 * NC + kc) << 9) + (lane << 4));
    v16bf b0 = load_frag_pair(pb0, pb0 + 1);
    acc0 = __builtin_amdgcn_wmma_f32_16x16x32_bf16(false, a[kc], false, b0,
                                                   (short)0, acc0, false, false);
  }
  float* g0 = gates + m0 * NGATE + col0;
#pragma unroll
  for (int r = 0; r < 8; ++r) g0[r * NGATE] = acc0[r];
}

// One layer: gates(16,400) = hA(16,K) * Wfrag(K,400) + bias, K = NC*32.
// Tile schedule (wave w): pair (w, w+8); then wave0: pair (16,24),
// waves 1..7: single (w+16). Branch is scalar (readfirstlane) => EXEC stays
// all-ones around every WMMA.
template <int NC>
__device__ __forceinline__ void wmma_layer(const __bf16* __restrict__ Wf,
                                           const __bf16* __restrict__ hA,
                                           const float* __restrict__ bias,
                                           float* __restrict__ gates,
                                           int lane, int wave) {
  const int am    = lane & 15;
  const int aBase = (lane & 16) ? 16 : 0;

  // hoisted A fragments (identical for every tile this wave owns)
  v16bf a[NC];
#pragma unroll
  for (int kc = 0; kc < NC; ++kc) {
    const H8* pa = reinterpret_cast<const H8*>(hA + am * K1PAD + kc * 32 + aBase);
    a[kc] = load_frag_pair(pa, pa + 1);   // +8 halves
  }

  gemm_pair<NC>(Wf, a, bias, gates, lane, wave, wave + 8);

  const int wsg = __builtin_amdgcn_readfirstlane(wave);   // scalar wave id
  if (wsg == 0) {
    gemm_pair<NC>(Wf, a, bias, gates, lane, 16, 24);
  } else {
    gemm_single<NC>(Wf, a, bias, gates, lane, wsg + 16);
  }
}

__global__ __launch_bounds__(256)
void lstm_ar_kernel(const float* __restrict__ xg,
                    const float* __restrict__ Wih0, const float* __restrict__ Whh0,
                    const float* __restrict__ bih0, const float* __restrict__ bhh0,
                    const float* __restrict__ Wih1, const float* __restrict__ Whh1,
                    const float* __restrict__ bih1, const float* __restrict__ bhh1,
                    const float* __restrict__ fcw_g, const float* __restrict__ fcb_g,
                    float* __restrict__ out) {
  extern __shared__ char smem[];
  __bf16* W0f  = reinterpret_cast<__bf16*>(smem + OFF_W0);
  __bf16* W1f  = reinterpret_cast<__bf16*>(smem + OFF_W1);
  __bf16* hA   = reinterpret_cast<__bf16*>(smem + OFF_HA);
  __bf16* xs   = reinterpret_cast<__bf16*>(smem + OFF_XS);
  float*  gts  = reinterpret_cast<float*>(smem + OFF_G);
  float*  b0f  = reinterpret_cast<float*>(smem + OFF_B0);
  float*  b1f  = reinterpret_cast<float*>(smem + OFF_B1);
  float*  wih  = reinterpret_cast<float*>(smem + OFF_WI);
  float*  fcwS = reinterpret_cast<float*>(smem + OFF_FC);
  float*  yv   = reinterpret_cast<float*>(smem + OFF_YV);

  const int tid  = threadIdx.x;
  const int lane = tid & 31;
  const int wave = tid >> 5;
  const int wgb  = blockIdx.x * ROWS;     // first batch row owned by this WG

  // ---- one-time init: pack weights into WMMA-fragment layout ----
  // Layer 0: B[k][n] = W_hh0[n][k], k padded 100->128 with zeros.
  for (int i = tid; i < NTILES * NC0 * 512; i += 256) {
    int q = i & 511, tc = i >> 9;
    int kc = tc & 3, tile = tc >> 2;
    int l = q >> 4, e = q & 15;
    int k = kc * 32 + e + ((l & 16) ? 16 : 0);
    int n = tile * 16 + (l & 15);
    float v = (k < HDIM) ? Whh0[n * HDIM + k] : 0.0f;
    W0f[i] = (__bf16)v;
  }
  // Layer 1: B[k][n] = k<100 ? W_ih1[n][k] : k<200 ? W_hh1[n][k-100] : 0.
  for (int i = tid; i < NTILES * NC1 * 512; i += 256) {
    int q = i & 511, tc = i >> 9;
    int kc = tc % NC1, tile = tc / NC1;
    int l = q >> 4, e = q & 15;
    int k = kc * 32 + e + ((l & 16) ? 16 : 0);
    int n = tile * 16 + (l & 15);
    float v = (k < HDIM) ? Wih1[n * HDIM + k]
            : (k < 2 * HDIM) ? Whh1[n * HDIM + (k - HDIM)] : 0.0f;
    W1f[i] = (__bf16)v;
  }
  for (int i = tid; i < ROWS * K1PAD; i += 256) hA[i] = (__bf16)0.0f;
  for (int i = tid; i < ROWS * SEQL; i += 256) {
    int m = i / SEQL, t = i - m * SEQL;
    xs[i] = (__bf16)xg[(wgb + m) * SEQL + t];
  }
  for (int i = tid; i < NGATE; i += 256) {
    b0f[i] = bih0[i] + bhh0[i];
    b1f[i] = bih1[i] + bhh1[i];
    wih[i] = Wih0[i];                      // W_ih0 is (400,1)
  }
  if (tid < HDIM) fcwS[tid] = fcw_g[tid];  // fc_w is (1,100)
  __syncthreads();

  // cell state in registers: thread tid owns cells {tid + 256*q}, cell=(m*100+j)
  float c0r[7] = {0.f, 0.f, 0.f, 0.f, 0.f, 0.f, 0.f};
  float c1r[7] = {0.f, 0.f, 0.f, 0.f, 0.f, 0.f, 0.f};
  const float fcb = fcb_g[0];

  for (int s = 0; s < NSTEPS; ++s) {
    for (int t = 0; t < SEQL; ++t) {
      // -------- layer 0 GEMM (+bias seed; W0f rows >=100 are zero) ---------
      wmma_layer<NC0>(W0f, hA, b0f, gts, lane, wave);
      __syncthreads();

      // -------- layer 0 elementwise ----------------------------------------
#pragma unroll
      for (int q = 0; q < 7; ++q) {
        int cell = tid + (q << 8);
        if (cell < ROWS * HDIM) {
          int m = cell / HDIM, j = cell - m * HDIM;
          float xt;
          if (s == 0)            xt = (float)xs[m * SEQL + t];
          else if (t < SEQL - 1) xt = (float)xs[m * SEQL + t + 1];
          else                   xt = yv[m];
          const float* gr = gts + m * NGATE;
          float ip = gr[j]       + wih[j]       * xt;
          float fp = gr[j + 100] + wih[j + 100] * xt;
          float gp = gr[j + 200] + wih[j + 200] * xt;
          float op = gr[j + 300] + wih[j + 300] * xt;
          float c  = fast_sig(fp) * c0r[q] + fast_sig(ip) * fast_tanh(gp);
          c0r[q]   = c;
          float h  = fast_sig(op) * fast_tanh(c);
          hA[m * K1PAD + j] = (__bf16)h;       // columns 0..99 = h0
        }
      }
      __syncthreads();

      // -------- layer 1 GEMM (A = [h0 | h1], +bias seed) -------------------
      wmma_layer<NC1>(W1f, hA, b1f, gts, lane, wave);
      __syncthreads();

      // -------- layer 1 elementwise ----------------------------------------
#pragma unroll
      for (int q = 0; q < 7; ++q) {
        int cell = tid + (q << 8);
        if (cell < ROWS * HDIM) {
          int m = cell / HDIM, j = cell - m * HDIM;
          float* gr = gts + m * NGATE;
          float ip = gr[j];
          float fp = gr[j + 100];
          float gp = gr[j + 200];
          float op = gr[j + 300];
          float c  = fast_sig(fp) * c1r[q] + fast_sig(ip) * fast_tanh(gp);
          c1r[q]   = c;
          float h  = fast_sig(op) * fast_tanh(c);
          hA[m * K1PAD + HDIM + j] = (__bf16)h; // columns 100..199 = h1
          if (t == SEQL - 1) gr[j] = h;         // stash f32 h1 for fc (reuse gates)
        }
      }
      __syncthreads();
    } // t

    // -------- fc head: y[m] = fc_b + h1[m,:] . fc_w ------------------------
    if (tid < ROWS) {
      float y = fcb;
      const float* hr = gts + tid * NGATE;
#pragma unroll 4
      for (int j = 0; j < HDIM; ++j) y += hr[j] * fcwS[j];
      yv[tid] = y;
      out[(wgb + tid) * NSTEPS + s] = y;
    }
    __syncthreads();
  } // s
}

extern "C" void kernel_launch(void* const* d_in, const int* in_sizes, int n_in,
                              void* d_out, int out_size, void* d_ws, size_t ws_size,
                              hipStream_t stream) {
  (void)in_sizes; (void)n_in; (void)out_size; (void)d_ws; (void)ws_size;
  const float* x    = (const float*)d_in[0];
  const float* Wih0 = (const float*)d_in[1];
  const float* Whh0 = (const float*)d_in[2];
  const float* bih0 = (const float*)d_in[3];
  const float* bhh0 = (const float*)d_in[4];
  const float* Wih1 = (const float*)d_in[5];
  const float* Whh1 = (const float*)d_in[6];
  const float* bih1 = (const float*)d_in[7];
  const float* bhh1 = (const float*)d_in[8];
  const float* fcw  = (const float*)d_in[9];
  const float* fcb  = (const float*)d_in[10];
  float* out = (float*)d_out;

  (void)hipFuncSetAttribute((const void*)lstm_ar_kernel,
                            hipFuncAttributeMaxDynamicSharedMemorySize, SMEM_BYTES);

  lstm_ar_kernel<<<BATCH / ROWS, 256, SMEM_BYTES, stream>>>(
      x, Wih0, Whh0, bih0, bhh0, Wih1, Whh1, bih1, bhh1, fcw, fcb, out);
}
